// MyLSTMCell_82703890252312
// MI455X (gfx1250) — matrix operand order repaired and checked by reading, not compiled, not run on hardware
//
#include <hip/hip_runtime.h>

// ---------------------------------------------------------------------------
// Fused LSTM cell for MI455X (gfx1250, wave32, WMMA).
//   gates = [x|h] @ Wcat^T  (M=65536, K=512, N=1024), fp16 WMMA, fp32 accum
//   then elementwise LSTM epilogue fully in registers.
// d_ws layout: [0, 1MB): fp16 weights pre-swizzled into WMMA B-fragment order
//              [1MB, 1MB+4KB): bias sums (bi + bh), fp32
// ---------------------------------------------------------------------------

typedef __attribute__((ext_vector_type(16))) _Float16     v16h;
typedef __attribute__((ext_vector_type(8)))  float        v8f;
typedef __attribute__((ext_vector_type(8)))  unsigned int v8u;

#define BATCH   65536
#define HDIM    256
#define KDIM    512            // I + H
#define BM      64             // batch rows per block
#define THREADS 512            // 16 waves
#define LDSROW  520            // halves per LDS row: 512 + 8 pad (16B) -> conflict-free frag reads

__device__ __forceinline__ float fsigmoid(float v) { return 1.0f / (1.0f + __expf(-v)); }
__device__ __forceinline__ float ftanh(float v)    { return 1.0f - 2.0f / (__expf(2.0f * v) + 1.0f); }

// ---------------------------------------------------------------------------
// Kernel 0: pack Wcat[n][k] (n in [0,1024), k in [0,512)) into fp16 fragments.
// Fragment (nt, kc): 16 N-cols x 32 K.  Per-lane 32B: lane l holds N=nt*16+l%16,
// VGPR j packs K pair {k0, k0+1}, k0 = kc*32 + 8*(l/16) + 2j + 8*(j/4).  Stored lane-major so the GEMM kernel
// reads each lane's 8 VGPRs as two contiguous b128 loads.
// Also computes bias_sum = bi + bh.
// ---------------------------------------------------------------------------
__global__ __launch_bounds__(256) void pack_weights(
    const float* __restrict__ Wi, const float* __restrict__ Wh,
    const float* __restrict__ bi, const float* __restrict__ bh,
    unsigned int* __restrict__ wpack, float* __restrict__ bias) {
  const int tid = blockIdx.x * 256 + threadIdx.x;      // 0 .. 262143
  const int j    =  tid        & 7;                    // VGPR index (K pair)
  const int lane = (tid >> 3)  & 31;                   // lane: N col (l%16), K half (l/16)
  const int kc   = (tid >> 8)  & 15;                   // K chunk
  const int nt   =  tid >> 12;                         // N tile (0..63)

  const int k  = kc * 32 + ((lane >> 4) << 3) + 2 * j + ((j >> 2) << 3);
  const int n0 = nt * 16 + (lane & 15);
  const int g  = n0 >> 8;                              // gate index
  const int hc = n0 & 255;                             // output column within gate

  float w0, w1;
  if (k < 256) {                                       // Wi part: [4][256][256]
    w0 = Wi[(g << 16) + (hc << 8) + k];
    w1 = Wi[(g << 16) + (hc << 8) + k + 1];
  } else {                                             // Wh part
    w0 = Wh[(g << 16) + (hc << 8) + (k - 256)];
    w1 = Wh[(g << 16) + (hc << 8) + (k - 255)];
  }
  const _Float16 h0 = (_Float16)w0, h1 = (_Float16)w1;
  wpack[tid] = (unsigned int)__builtin_bit_cast(unsigned short, h0)
             | ((unsigned int)__builtin_bit_cast(unsigned short, h1) << 16);

  if (tid < 1024) bias[tid] = bi[tid] + bh[tid];
}

// ---------------------------------------------------------------------------
// Kernel 1: fused GEMM + LSTM epilogue.
// Block: 64 rows x 1024 gate-cols.  Wave w owns gate-col tile w (16 cols of H)
// for all 4 gates -> epilogue is in-register.
// ---------------------------------------------------------------------------
__global__ __launch_bounds__(THREADS) void lstm_fused(
    const float* __restrict__ x, const float* __restrict__ h,
    const float* __restrict__ c, const unsigned int* __restrict__ wpack,
    const float* __restrict__ bias, float* __restrict__ out) {
  __shared__ __align__(16) unsigned short Atile[BM][LDSROW];

  const int tid     = threadIdx.x;
  const int rowBase = blockIdx.x * BM;

  // ---- stage A = [x | h] tile as fp16 in LDS (coalesced float4 reads) ----
#pragma unroll
  for (int it = 0; it < (BM * KDIM / 4) / THREADS; ++it) {   // 16 iterations
    const int e   = it * THREADS + tid;
    const int row = e >> 7;                                  // 128 float4 per row
    const int k4  = (e & 127) << 2;                          // k offset (mult of 4)
    const float* src = (k4 < 256) ? (x + (rowBase + row) * HDIM + k4)
                                  : (h + (rowBase + row) * HDIM + (k4 - 256));
    const float4 v = *(const float4*)src;
    const _Float16 f0 = (_Float16)v.x, f1 = (_Float16)v.y,
                   f2 = (_Float16)v.z, f3 = (_Float16)v.w;
    const unsigned int p0 = (unsigned int)__builtin_bit_cast(unsigned short, f0)
                          | ((unsigned int)__builtin_bit_cast(unsigned short, f1) << 16);
    const unsigned int p1 = (unsigned int)__builtin_bit_cast(unsigned short, f2)
                          | ((unsigned int)__builtin_bit_cast(unsigned short, f3) << 16);
    *(uint2*)&Atile[row][k4] = uint2{p0, p1};
  }
  __syncthreads();

  const int wave   = tid >> 5;        // 0..15 : gate-column tile
  const int lane   = tid & 31;
  const int laneLo = lane & 15;
  const int laneHi = lane >> 4;       // selects K sub-half of A fragment

  v8f acc[4][4] = {};                 // [row-tile][gate], fp32 accumulators

  for (int kc = 0; kc < 16; ++kc) {   // K chunks of 32
    // A fragments (16x32 fp16) for the 4 row tiles, from LDS.
    // lane<16: row=laneLo, K {0..7,16..23}; lane>=16: same row, K {8..15,24..31}
    union { uint4 u[2]; v16h v; } a[4];
#pragma unroll
    for (int r = 0; r < 4; ++r) {
      const unsigned short* ap = &Atile[r * 16 + laneLo][0] + kc * 32 + laneHi * 8;
      a[r].u[0] = *(const uint4*)ap;
      a[r].u[1] = *(const uint4*)(ap + 16);
    }
#pragma unroll
    for (int g = 0; g < 4; ++g) {
      // B fragment (32x16 fp16) for gate g, gate-col tile `wave`: prepacked, L2-resident
      const unsigned int* bp = wpack + ((((g * 16 + wave) * 16 + kc) << 8) + lane * 8);
      const v16h bf = __builtin_bit_cast(v16h, *(const v8u*)bp);
#pragma unroll
      for (int r = 0; r < 4; ++r)
        acc[r][g] = __builtin_amdgcn_wmma_f32_16x16x32_f16(
            false, a[r].v, false, bf, (short)0, acc[r][g], false, false);
    }
  }

  // ---- epilogue: bias + activations + state update, all in registers ----
  // C/D layout: element (m,n): lane = n + 16*(m/8), VGPR j = m%8
  const int col = (wave << 4) + laneLo;
  const float b0 = bias[0 * HDIM + col];
  const float b1 = bias[1 * HDIM + col];
  const float b2 = bias[2 * HDIM + col];
  const float b3 = bias[3 * HDIM + col];
  const int HN = BATCH * HDIM;        // 16,777,216 (< 2^31 elements total)

#pragma unroll
  for (int r = 0; r < 4; ++r) {
    const int row0 = rowBase + r * 16 + laneHi * 8;
#pragma unroll
    for (int j = 0; j < 8; ++j) {
      const int idx = (row0 + j) * HDIM + col;
      const float ig = fsigmoid(acc[r][0][j] + b0);
      const float fg = fsigmoid(acc[r][1][j] + b1);
      const float gg = ftanh   (acc[r][2][j] + b2);
      const float og = fsigmoid(acc[r][3][j] + b3);
      const float cp = c[idx];
      const float cn = fg * cp + ig * gg;
      const float hn = og * ftanh(cn);
      out[idx]      = hn;             // h_new (first tuple element)
      out[HN + idx] = cn;             // c_new (second tuple element)
    }
  }
}

// ---------------------------------------------------------------------------
extern "C" void kernel_launch(void* const* d_in, const int* in_sizes, int n_in,
                              void* d_out, int out_size, void* d_ws, size_t ws_size,
                              hipStream_t stream) {
  (void)in_sizes; (void)n_in; (void)out_size; (void)ws_size;
  const float* x  = (const float*)d_in[0];
  const float* h  = (const float*)d_in[1];
  const float* c  = (const float*)d_in[2];
  const float* Wi = (const float*)d_in[3];
  const float* Wh = (const float*)d_in[4];
  const float* bi = (const float*)d_in[5];
  const float* bh = (const float*)d_in[6];

  unsigned int* wpack = (unsigned int*)d_ws;                 // 1 MB fp16 fragments
  float*        bias  = (float*)((char*)d_ws + (1 << 20));   // 4 KB bias sums

  pack_weights<<<1024, 256, 0, stream>>>(Wi, Wh, bi, bh, wpack, bias);
  lstm_fused <<<BATCH / BM, THREADS, 0, stream>>>(x, h, c, wpack, bias, (float*)d_out);
}
